// MultipleMLP_61400852464330
// MI455X (gfx1250) — matrix operand order repaired and checked by reading, not compiled, not run on hardware
//
#include <hip/hip_runtime.h>

#define B_DIM 256
#define C_DIM 4096
#define FIN   160
#define FHID  192
#define EPSV  1e-5f

typedef __attribute__((ext_vector_type(16))) __bf16 v16bf;
typedef __attribute__((ext_vector_type(4)))  __bf16 v4bf;
typedef __attribute__((ext_vector_type(2)))  __bf16 v2bf;
typedef __attribute__((ext_vector_type(16))) float  v16f;
typedef __attribute__((ext_vector_type(4)))  float  v4f;
typedef __attribute__((ext_vector_type(8)))  float  v8f;
typedef __attribute__((ext_vector_type(4)))  unsigned int v4u;

union ABfrag { v16bf v; v4u u[2]; };

__device__ __forceinline__ float bf2f(unsigned int h) {
  return __uint_as_float(h << 16);
}

// LDS budget: 61440 (W1 bf16) + 98304 (h bf16, XOR-swizzled) + 3840 (coeffs)
// = 163584 B <= 160 KB  ==>  2 workgroups per WGP: one WG streams HBM while
// the other runs its WMMA phase, hiding the GEMM under the 23.3 TB/s stream.
// __launch_bounds__(256, 4): guarantee 4 waves/SIMD (16 waves/WGP) fit the
// register file, i.e. cap per-wave VGPRs so the 2-WG occupancy is achievable.
__global__ __launch_bounds__(256, 4) void mlp_cell_kernel(
    const float* __restrict__ x, const float* __restrict__ W1,
    const float* __restrict__ gamma, const float* __restrict__ beta,
    const float* __restrict__ W2, const float* __restrict__ b2,
    float* __restrict__ out)
{
  __shared__ __align__(16) unsigned short ws[FHID * FIN];    // 60 KB bf16 W1_c
  __shared__ __align__(16) unsigned short hs[B_DIM * FHID];  // 96 KB bf16 h (swizzled)
  __shared__ float ssum[FHID], ssq[FHID];
  __shared__ float sscale[FHID], sshift[FHID], sw2[FHID];

  const int c = blockIdx.x;
  const int t = threadIdx.x;

  // ---------------- Phase 0: stage W1_c into LDS as bf16 (coalesced) -------
  {
    const float* wg = W1 + (size_t)c * (FHID * FIN);  // contiguous per cell
    for (int j = 0; j < 30; ++j) {
      int e = (j * 256 + t) * 4;
      v4f vf = *(const v4f*)(wg + e);
      v4bf vb = __builtin_convertvector(vf, v4bf);    // native fptrunc (RNE)
      *(unsigned long long*)(&ws[e]) = __builtin_bit_cast(unsigned long long, vb);
    }
    if (t < FHID) { ssum[t] = 0.0f; ssq[t] = 0.0f; }
  }
  __syncthreads();

  // ---------------- Phase 1: h = x @ W1^T via WMMA bf16 --------------------
  // M = 256 (batch), N = 192 (hidden), K = 160 = 5 * 32.
  // 8 waves; each wave computes 2 M-tiles (wave, wave+8).
  // A-fragments gathered straight from global fp32 x (16B chunks, HBM-once);
  // B-fragments double-buffered from LDS so >=12 ds_loads stay in flight
  // under the WMMA stream (graduated s_wait_dscnt instead of dscnt==0).
  const int wave = t >> 5;
  const int lane = t & 31;
  const int lq   = lane & 15;       // N (and A-row) selector within tile
  const int lhi  = lane >> 4;       // half-wave selector

  for (int half = 0; half < 2; ++half) {
    const int mt = wave + half * 8;          // M-tile index 0..15
    const int m  = mt * 16 + lq;             // this lane's A row

    // ---- load + convert all 5 A fragments for this M-tile (40 VGPRs) ----
    ABfrag A[5];
    {
      const float* xrow = x + ((size_t)m * C_DIM + c) * FIN;
      #pragma unroll
      for (int kt = 0; kt < 5; ++kt) {
        const int ka = kt * 32 + lhi * 8;    // lanes 0-15: K=[0..7],[16..23];
        v4f f0 = *(const v4f*)(xrow + ka);             // 16-31: [8..15],[24..31]
        v4f f1 = *(const v4f*)(xrow + ka + 4);
        v4f f2 = *(const v4f*)(xrow + ka + 16);
        v4f f3 = *(const v4f*)(xrow + ka + 20);
        v16f xf = {f0[0],f0[1],f0[2],f0[3], f1[0],f1[1],f1[2],f1[3],
                   f2[0],f2[1],f2[2],f2[3], f3[0],f3[1],f3[2],f3[3]};
        A[kt].v = __builtin_convertvector(xf, v16bf);  // v_cvt_pk_bf16_f32 x8
      }
    }

    for (int pass = 0; pass < 2; ++pass) {
      // Single base pointer; every B load below becomes base + const offset
      // (q*5120 + kt*64 (+32) bytes, all < 64 KiB -> DS immediate offsets).
      const unsigned short* bbase = &ws[(pass * 96 + lq) * FIN + lhi * 8];

      v8f acc[6];
      #pragma unroll
      for (int q = 0; q < 6; ++q) {
        v8f z = {0.f,0.f,0.f,0.f,0.f,0.f,0.f,0.f};
        acc[q] = z;
      }

      ABfrag Bb[2][6];
      #pragma unroll
      for (int q = 0; q < 6; ++q) {          // prime the pipe: kt = 0
        Bb[0][q].u[0] = *(const v4u*)(bbase + q * (16 * FIN));
        Bb[0][q].u[1] = *(const v4u*)(bbase + q * (16 * FIN) + 16);
      }

      #pragma unroll
      for (int kt = 0; kt < 5; ++kt) {
        const int cur = kt & 1, nxt = cur ^ 1;
        if (kt < 4) {                        // prefetch kt+1 before using kt
          #pragma unroll
          for (int q = 0; q < 6; ++q) {
            Bb[nxt][q].u[0] = *(const v4u*)(bbase + q * (16 * FIN) + (kt + 1) * 32);
            Bb[nxt][q].u[1] = *(const v4u*)(bbase + q * (16 * FIN) + (kt + 1) * 32 + 16);
          }
        }
        #pragma unroll
        for (int q = 0; q < 6; ++q) {
          acc[q] = __builtin_amdgcn_wmma_f32_16x16x32_bf16(
              false, A[kt].v, false, Bb[cur][q].v, (short)0, acc[q], false, false);
        }
      }

      // Emit h (bf16, XOR bank-swizzled rows) + fp32 batch statistics.
      // C/D layout: VGPR r, lanes 0-15 -> (M=r, N=lq); lanes 16-31 -> (M=8+r, N=lq)
      #pragma unroll
      for (int q = 0; q < 6; ++q) {
        const int n = (pass * 6 + q) * 16 + lq;
        const int p = n >> 1;                // dword-pair column
        const int nl = n & 1;
        float s = 0.0f, sq = 0.0f;
        #pragma unroll
        for (int r = 0; r < 8; ++r) {
          float vv = acc[q][r];
          s  += vv;
          sq += vv * vv;
          const int m2 = mt * 16 + lhi * 8 + r;
          hs[m2 * FHID + (((p ^ (m2 & 31)) << 1) | nl)] =
              __builtin_bit_cast(unsigned short, (__bf16)vv);
        }
        atomicAdd(&ssum[n], s);   // ds_add_f32
        atomicAdd(&ssq[n],  sq);
      }
    }
  }
  __syncthreads();

  // ---------------- Phase 2: batchnorm coefficients ------------------------
  if (t < FHID) {
    float mean = ssum[t] * (1.0f / 256.0f);
    float var  = ssq[t]  * (1.0f / 256.0f) - mean * mean;   // biased variance
    float sc   = gamma[(size_t)c * FHID + t] * rsqrtf(var + EPSV);
    sscale[t]  = sc;
    sshift[t]  = beta[(size_t)c * FHID + t] - mean * sc;
    sw2[t]     = W2[(size_t)c * FHID + t];
  }
  __syncthreads();

  // ---------------- Phase 3: normalize + ReLU + dot(W2) --------------------
  // Thread t owns batch row t; XOR de-swizzle keeps all 32 lanes of a wave on
  // distinct LDS banks for every iteration.
  {
    float acc2 = 0.0f;
    const unsigned int* hrow = (const unsigned int*)(&hs[t * FHID]);
    const int sw = t & 31;
    #pragma unroll 4
    for (int fp = 0; fp < FHID / 2; ++fp) {
      unsigned int pkv = hrow[fp ^ sw];      // 2 bf16 for features 2fp, 2fp+1
      int f = fp * 2;
      float v0 = fmaf(bf2f(pkv & 0xffffu), sscale[f],     sshift[f]);
      float v1 = fmaf(bf2f(pkv >> 16),     sscale[f + 1], sshift[f + 1]);
      v0 = fmaxf(v0, 0.0f);
      v1 = fmaxf(v1, 0.0f);
      acc2 = fmaf(v0, sw2[f],     acc2);
      acc2 = fmaf(v1, sw2[f + 1], acc2);
    }
    out[(size_t)t * C_DIM + c] = acc2 + b2[c];
  }
}

extern "C" void kernel_launch(void* const* d_in, const int* in_sizes, int n_in,
                              void* d_out, int out_size, void* d_ws, size_t ws_size,
                              hipStream_t stream) {
  (void)in_sizes; (void)n_in; (void)out_size; (void)d_ws; (void)ws_size;
  const float* x     = (const float*)d_in[0];
  const float* W1    = (const float*)d_in[1];
  const float* gamma = (const float*)d_in[2];
  const float* beta  = (const float*)d_in[3];
  const float* W2    = (const float*)d_in[4];
  const float* b2    = (const float*)d_in[5];
  float* out = (float*)d_out;

  mlp_cell_kernel<<<C_DIM, 256, 0, stream>>>(x, W1, gamma, beta, W2, b2, out);
}
